// SelfJoinLayer_25709674234349
// MI455X (gfx1250) — matrix operand order repaired and checked by reading, not compiled, not run on hardware
//
#include <hip/hip_runtime.h>
#include <hip/hip_bf16.h>

// Problem constants (match reference)
#define CDIM   128
#define NSROWS 8192
#define MROWS  8192
#define NTROWS 10000
#define TOPK   20

typedef __attribute__((ext_vector_type(16))) _Float16 v16h;
typedef __attribute__((ext_vector_type(8)))  _Float16 v8h;
typedef __attribute__((ext_vector_type(8)))  float    v8f;
typedef __attribute__((ext_vector_type(4)))  unsigned int u32x4;
typedef __attribute__((ext_vector_type(4)))  int          i32x4;
typedef __attribute__((ext_vector_type(8)))  int          i32x8;

#ifndef __has_builtin
#define __has_builtin(x) 0
#endif
#if defined(__AMDGCN__) && __has_builtin(__builtin_amdgcn_tensor_load_to_lds) && \
    __has_builtin(__builtin_amdgcn_s_wait_tensorcnt)
#define USE_TDM 1
#else
#define USE_TDM 0
#endif

// ---------------------------------------------------------------------------
// WMMA fragment loaders (CDNA5 16x16x32 f16, wave32 layouts per ISA 7.12.2)
// A: 16x32 (MxK).  lane<16: row=lane,   K = {k0..k0+7, k0+16..k0+23}
//                  lane>=16: row=lane-16, K = {k0+8..k0+15, k0+24..k0+31}
// B: 32x16 (KxN) from Bt[n, k] (row-major by n):
//                  lane<16:  n=n0+lane,    K = {k0..k0+7, k0+16..k0+23}
//                  lane>=16: n=n0+lane-16, K = {k0+8..k0+15, k0+24..k0+31}
// ---------------------------------------------------------------------------
__device__ __forceinline__ v16h load_a_frag(const _Float16* __restrict__ A,
                                            int row0, int k0) {
  const int l = threadIdx.x & 31;
  const _Float16* p = A + (size_t)(row0 + (l & 15)) * CDIM + k0 + ((l < 16) ? 0 : 8);
  union { v16h v; v8h h[2]; } u;
  u.h[0] = *(const v8h*)(p);
  u.h[1] = *(const v8h*)(p + 16);
  return u.v;
}

__device__ __forceinline__ v16h load_b_frag(const _Float16* __restrict__ Bt,
                                            int n0, int k0, int ldk) {
  const int l = threadIdx.x & 31;
  const _Float16* p = Bt + (size_t)(n0 + (l & 15)) * ldk + k0 + ((l < 16) ? 0 : 8);
  union { v16h v; v8h h[2]; } u;
  u.h[0] = *(const v8h*)(p);
  u.h[1] = *(const v8h*)(p + 16);
  return u.v;
}

// ---------------------------------------------------------------------------
// Generic M x 128 = (M x 128) @ (128 x 128) WMMA GEMM, epilogue templated so
// all flag branches constant-fold (straight-line stores, no scalar ladders).
// block = 256 threads (8 waves); wave w computes n-tile w; grid.x = M/16.
// ---------------------------------------------------------------------------
template <bool HAS_BIAS, bool HAS_ROWSCALE, bool HAS_D, bool RELU,
          bool OUT_F, bool OUT_H>
__global__ void gemm128_kernel(const _Float16* __restrict__ A,
                               const _Float16* __restrict__ Bt,
                               const float* __restrict__ bias,
                               const float* __restrict__ rowScale,
                               const float* __restrict__ Dadd,
                               float* __restrict__ outF,
                               _Float16* __restrict__ outH) {
  const int row0 = blockIdx.x * 16;
  const int n0 = (threadIdx.x >> 5) * 16;
  v8f c = {};
#pragma unroll
  for (int k0 = 0; k0 < CDIM; k0 += 32) {
    v16h a = load_a_frag(A, row0, k0);
    v16h b = load_b_frag(Bt, n0, k0, CDIM);
    c = __builtin_amdgcn_wmma_f32_16x16x32_f16(false, a, false, b,
                                               (short)0, c, false, false);
  }
  const int l = threadIdx.x & 31;
  const int n = n0 + (l & 15);
  const int rbase = (l < 16) ? 0 : 8;
  const float bn = HAS_BIAS ? bias[n] : 0.0f;
#pragma unroll
  for (int r = 0; r < 8; ++r) {
    const int m = row0 + rbase + r;
    float v = c[r];
    if (HAS_BIAS) v += (HAS_ROWSCALE ? rowScale[m] : 1.0f) * bn;
    if (HAS_D) v += Dadd[(size_t)m * CDIM + n];
    if (RELU) v = fmaxf(v, 0.0f);
    const size_t off = (size_t)m * CDIM + n;
    if (OUT_F) outF[off] = v;
    if (OUT_H) outH[off] = (_Float16)v;
  }
}

// ---------------------------------------------------------------------------
// Stage a 128x128 f16 strip of Q (rows j0..j0+127) into LDS.
// TDM path: one wave issues a 2D tensor_load_to_lds D# (data_size=2B,
// tile 128x128, tensor stride 128) and waits on TENSORcnt. Fallback:
// cooperative vectorized global->LDS copy.
// ---------------------------------------------------------------------------
__device__ __forceinline__ void stage_q_tile(_Float16* qTile,
                                             const _Float16* __restrict__ Qh,
                                             int j0) {
#if USE_TDM
  if ((threadIdx.x >> 5) == 0) {
    const unsigned long long ga =
        (unsigned long long)(const void*)(Qh + (size_t)j0 * CDIM);
    const unsigned ldsOff = (unsigned)(size_t)(void*)qTile;
    // D# group 0: count=1 | lds_addr | global_addr(57b) | type=2 ("image")
    u32x4 g0;
    g0.x = 1u;                                   // count=1, user mode
    g0.y = ldsOff;                               // lds_addr (bytes)
    g0.z = (unsigned)(ga & 0xFFFFFFFFu);         // global_addr[31:0]
    g0.w = (unsigned)((ga >> 32) & 0x01FFFFFFu)  // global_addr[56:32]
           | (2u << 30);                         // type = 2
    // D# group 1: data_size=2B; tensor 128 x 8192; tile 128x128; stride0=128
    i32x8 g1;
    g1[0] = (int)(1u << 16);                     // data_size=1 -> 2 bytes
    g1[1] = (int)((128u & 0xFFFFu) << 16);       // tensor_dim0 lo16 @bit48
    g1[2] = (int)(((128u >> 16) & 0xFFFFu)       // tensor_dim0 hi16
                  | ((8192u & 0xFFFFu) << 16));  // tensor_dim1 lo16
    g1[3] = (int)(((8192u >> 16) & 0xFFFFu)      // tensor_dim1 hi16
                  | (128u << 16));               // tile_dim0 = 128
    g1[4] = (int)(128u);                         // tile_dim1=128, tile_dim2=0
    g1[5] = (int)(128u);                         // tensor_dim0_stride lo32
    g1[6] = 0;                                   // stride0 hi16 | stride1 lo16
    g1[7] = 0;                                   // stride1 hi32
    i32x4 gz = {0, 0, 0, 0};                     // groups 2/3 unused (2D tile)
#if __clang_major__ >= 23
    i32x8 gz8 = {0, 0, 0, 0, 0, 0, 0, 0};
    __builtin_amdgcn_tensor_load_to_lds(g0, g1, gz, gz, gz8, 0);
#else
    __builtin_amdgcn_tensor_load_to_lds(g0, g1, gz, gz, 0);
#endif
    __builtin_amdgcn_s_wait_tensorcnt(0);
  }
#else
  // Cooperative copy: 128*128 halves = 2048 x v8h; 256 threads x 8 each.
  const _Float16* src = Qh + (size_t)j0 * CDIM;
  for (int e = threadIdx.x; e < (128 * 128) / 8; e += 256) {
    ((v8h*)qTile)[e] = ((const v8h*)src)[e];
  }
#endif
}

// ---------------------------------------------------------------------------
// sim = Kh @ Qh^T with time-mask, streaming top-20 + softmax per row.
// block = 256 threads (8 waves) owns 16 rows of Kh; per iteration the block
// stages a 128-column Q strip into LDS (TDM), then each wave computes one
// 16x16 WMMA tile (K=128 in 4 chunks) with B fragments read from LDS.
// ---------------------------------------------------------------------------
__global__ void sim_topk_kernel(const _Float16* __restrict__ Kh,
                                const _Float16* __restrict__ Qh,
                                const int* __restrict__ st,
                                const int* __restrict__ bst,
                                int* __restrict__ topIdx,
                                float* __restrict__ topScore) {
  __shared__ _Float16 qTile[128 * 128];   // 32 KB staged Q strip
  __shared__ float tileS[16 * 128];
  __shared__ float topv[16][TOPK];
  __shared__ int   topi[16][TOPK];
  __shared__ int   strow[16];

  const int tid = threadIdx.x;
  const int wave = tid >> 5;
  const int lane = tid & 31;
  const int i0 = blockIdx.x * 16;
  const float NEGINF = -__builtin_inff();

  for (int e = tid; e < 16 * TOPK; e += 256) {
    topv[e / TOPK][e % TOPK] = NEGINF;
    topi[e / TOPK][e % TOPK] = 0;
  }
  if (tid < 16) strow[tid] = st[i0 + tid];

  // Persistent A fragments: 16 rows x K=128 of Kh
  v16h a0 = load_a_frag(Kh, i0, 0);
  v16h a1 = load_a_frag(Kh, i0, 32);
  v16h a2 = load_a_frag(Kh, i0, 64);
  v16h a3 = load_a_frag(Kh, i0, 96);

  float mn = NEGINF;   // row-owner cached min of top list
  int   mp = 0;

  for (int j0 = 0; j0 < MROWS; j0 += 128) {
    __syncthreads();                       // qTile/tileS free to overwrite
    stage_q_tile(qTile, Qh, j0);
    __syncthreads();                       // staged strip visible to all

    const int nloc = wave * 16;            // this wave's local n-tile in strip
    v8f c = {};
    c = __builtin_amdgcn_wmma_f32_16x16x32_f16(false, a0, false, load_b_frag(qTile, nloc, 0, CDIM),  (short)0, c, false, false);
    c = __builtin_amdgcn_wmma_f32_16x16x32_f16(false, a1, false, load_b_frag(qTile, nloc, 32, CDIM), (short)0, c, false, false);
    c = __builtin_amdgcn_wmma_f32_16x16x32_f16(false, a2, false, load_b_frag(qTile, nloc, 64, CDIM), (short)0, c, false, false);
    c = __builtin_amdgcn_wmma_f32_16x16x32_f16(false, a3, false, load_b_frag(qTile, nloc, 96, CDIM), (short)0, c, false, false);

    const int n = lane & 15;
    const int bj = bst[j0 + nloc + n];
    const int rbase = (lane < 16) ? 0 : 8;
#pragma unroll
    for (int r = 0; r < 8; ++r) {
      float v = c[r];
      if (!(strow[rbase + r] < bj)) v = NEGINF;   // mask invalid pairs
      tileS[(rbase + r) * 128 + nloc + n] = v;
    }
    __syncthreads();

    if (tid < 16) {  // row owner scans its 128 candidates
      const float* rowp = &tileS[tid * 128];
      for (int jj = 0; jj < 128; ++jj) {
        const float v = rowp[jj];
        if (v > mn) {
          topv[tid][mp] = v;
          topi[tid][mp] = j0 + jj;
          mn = topv[tid][0]; mp = 0;
#pragma unroll
          for (int s = 1; s < TOPK; ++s) {
            const float t = topv[tid][s];
            if (t < mn) { mn = t; mp = s; }
          }
        }
      }
    }
  }
  __syncthreads();

  if (tid < 16) {  // softmax over top-20 (all -inf -> scores 0, per nan_to_num)
    const int i = i0 + tid;
    float mx = NEGINF;
#pragma unroll
    for (int s = 0; s < TOPK; ++s) mx = fmaxf(mx, topv[tid][s]);
    float ev[TOPK];
    float esum = 0.0f;
#pragma unroll
    for (int s = 0; s < TOPK; ++s) {
      const float v = topv[tid][s];
      const float e = (mx > NEGINF && v > NEGINF) ? __expf(v - mx) : 0.0f;
      ev[s] = e;
      esum += e;
    }
    const float inv = (esum > 0.0f) ? 1.0f / esum : 0.0f;
#pragma unroll
    for (int s = 0; s < TOPK; ++s) {
      topScore[(size_t)i * TOPK + s] = ev[s] * inv;
      topIdx[(size_t)i * TOPK + s] = topi[tid][s];
    }
  }
}

// ---------------------------------------------------------------------------
// G[i] = sum_s score * relu(U[i] + V[j_s] + y_{j_s}*w1y)   (msg_b1 is in U)
// ssum[i] = sum_s score   (scales msg_b2 in the following GEMM)
// ---------------------------------------------------------------------------
__global__ void msg_agg_kernel(const float* __restrict__ U,
                               const float* __restrict__ V,
                               const float* __restrict__ bank_y,
                               const float* __restrict__ w1y,
                               const int* __restrict__ topIdx,
                               const float* __restrict__ topScore,
                               _Float16* __restrict__ Gh,
                               float* __restrict__ ssum) {
  const int i = blockIdx.x;
  const int ch = threadIdx.x;
  __shared__ int   sj[TOPK];
  __shared__ float ssc[TOPK];
  __shared__ float sy[TOPK];
  if (ch < TOPK) {
    const int j = topIdx[(size_t)i * TOPK + ch];
    sj[ch] = j;
    ssc[ch] = topScore[(size_t)i * TOPK + ch];
    sy[ch] = bank_y[j];
  }
  __syncthreads();
  const float u = U[(size_t)i * CDIM + ch];
  const float wy = w1y[ch];
  float acc = 0.0f;
#pragma unroll
  for (int s = 0; s < TOPK; ++s) {
    const float t = u + V[(size_t)sj[s] * CDIM + ch] + sy[s] * wy;
    acc += ssc[s] * fmaxf(t, 0.0f);
  }
  Gh[(size_t)i * CDIM + ch] = (_Float16)acc;
  if (ch == 0) {
    float s2 = 0.0f;
#pragma unroll
    for (int s = 0; s < TOPK; ++s) s2 += ssc[s];
    ssum[i] = s2;
  }
}

// ---------------------------------------------------------------------------
// Helpers: convert / transpose / copy
// ---------------------------------------------------------------------------
__global__ void f32_to_f16_kernel(const float* __restrict__ src,
                                  _Float16* __restrict__ dst, int n) {
  const int i = blockIdx.x * 256 + threadIdx.x;
  if (i < n) dst[i] = (_Float16)src[i];
}

__global__ void copy_f32_kernel(const float* __restrict__ src,
                                float* __restrict__ dst, int n) {
  const int i = blockIdx.x * 256 + threadIdx.x;
  if (i < n) dst[i] = src[i];
}

// Wt[n*128+k] = W[(rowOff+k)*128 + n]   (128x128 slab); grid=<128,128>
__global__ void transpose_w_kernel(const float* __restrict__ W,
                                   _Float16* __restrict__ Wt, int rowOff) {
  const int n = blockIdx.x;
  const int k = threadIdx.x;
  Wt[(size_t)n * CDIM + k] = (_Float16)W[(size_t)(rowOff + k) * CDIM + n];
}

// ---------------------------------------------------------------------------
extern "C" void kernel_launch(void* const* d_in, const int* in_sizes, int n_in,
                              void* d_out, int out_size, void* d_ws, size_t ws_size,
                              hipStream_t stream) {
  (void)in_sizes; (void)n_in; (void)out_size; (void)ws_size;

  const float* x        = (const float*)d_in[0];
  const float* bank_x   = (const float*)d_in[1];
  const float* bank_y   = (const float*)d_in[2];
  const int*   st       = (const int*)d_in[3];
  const int*   bst      = (const int*)d_in[4];
  const float* Wq       = (const float*)d_in[5];
  const float* bq       = (const float*)d_in[6];
  const float* Wk       = (const float*)d_in[7];
  const float* bk       = (const float*)d_in[8];
  const float* msg_W1   = (const float*)d_in[9];
  const float* msg_b1   = (const float*)d_in[10];
  const float* msg_W2   = (const float*)d_in[11];
  const float* msg_b2   = (const float*)d_in[12];
  const float* upd_W1   = (const float*)d_in[13];
  const float* upd_b1   = (const float*)d_in[14];
  const float* upd_W2   = (const float*)d_in[15];
  const float* upd_b2   = (const float*)d_in[16];
  float* out = (float*)d_out;

  // ---- workspace carve-up (~29 MB) ----
  char* ws = (char*)d_ws;
  size_t off = 0;
  auto alloc = [&](size_t bytes) {
    void* p = ws + off;
    off += (bytes + 255) & ~(size_t)255;
    return p;
  };
  _Float16* feat_h = (_Float16*)alloc((size_t)NSROWS * CDIM * 2);
  _Float16* bank_h = (_Float16*)alloc((size_t)MROWS * CDIM * 2);
  _Float16* q_h    = (_Float16*)alloc((size_t)MROWS * CDIM * 2);
  _Float16* k_h    = (_Float16*)alloc((size_t)NSROWS * CDIM * 2);
  _Float16* Wq_t   = (_Float16*)alloc((size_t)CDIM * CDIM * 2);
  _Float16* Wk_t   = (_Float16*)alloc((size_t)CDIM * CDIM * 2);
  _Float16* W1a_t  = (_Float16*)alloc((size_t)CDIM * CDIM * 2);
  _Float16* W1b_t  = (_Float16*)alloc((size_t)CDIM * CDIM * 2);
  _Float16* W2m_t  = (_Float16*)alloc((size_t)CDIM * CDIM * 2);
  _Float16* U1a_t  = (_Float16*)alloc((size_t)CDIM * CDIM * 2);
  _Float16* U1b_t  = (_Float16*)alloc((size_t)CDIM * CDIM * 2);
  _Float16* W2u_t  = (_Float16*)alloc((size_t)CDIM * CDIM * 2);
  float*    w1y    = (float*)alloc((size_t)CDIM * 4);
  float*    U      = (float*)alloc((size_t)NSROWS * CDIM * 4);
  float*    V      = (float*)alloc((size_t)MROWS * CDIM * 4);
  float*    P      = (float*)alloc((size_t)NSROWS * CDIM * 4);
  int*      topIdx = (int*)alloc((size_t)NSROWS * TOPK * 4);
  float*    topScr = (float*)alloc((size_t)NSROWS * TOPK * 4);
  _Float16* G_h    = (_Float16*)alloc((size_t)NSROWS * CDIM * 2);
  float*    ssum   = (float*)alloc((size_t)NSROWS * 4);
  _Float16* H_h    = (_Float16*)alloc((size_t)NSROWS * CDIM * 2);
  _Float16* R1_h   = (_Float16*)alloc((size_t)NSROWS * CDIM * 2);

  // ---- stage 0: precision conversion + weight transposes ----
  {
    const int n1 = NSROWS * CDIM, n2 = MROWS * CDIM;
    f32_to_f16_kernel<<<(n1 + 255) / 256, 256, 0, stream>>>(x, feat_h, n1);
    f32_to_f16_kernel<<<(n2 + 255) / 256, 256, 0, stream>>>(bank_x, bank_h, n2);
  }
  transpose_w_kernel<<<CDIM, CDIM, 0, stream>>>(Wq, Wq_t, 0);
  transpose_w_kernel<<<CDIM, CDIM, 0, stream>>>(Wk, Wk_t, 0);
  transpose_w_kernel<<<CDIM, CDIM, 0, stream>>>(msg_W1, W1a_t, 0);
  transpose_w_kernel<<<CDIM, CDIM, 0, stream>>>(msg_W1, W1b_t, CDIM);
  transpose_w_kernel<<<CDIM, CDIM, 0, stream>>>(msg_W2, W2m_t, 0);
  transpose_w_kernel<<<CDIM, CDIM, 0, stream>>>(upd_W1, U1a_t, 0);
  transpose_w_kernel<<<CDIM, CDIM, 0, stream>>>(upd_W1, U1b_t, CDIM);
  transpose_w_kernel<<<CDIM, CDIM, 0, stream>>>(upd_W2, W2u_t, 0);
  copy_f32_kernel<<<1, 256, 0, stream>>>(msg_W1 + (size_t)(2 * CDIM) * CDIM, w1y, CDIM);

  // ---- stage 1: projection + MLP-input GEMMs (all WMMA) ----
  // q = bank @ Wq + bq  (f16 out)
  gemm128_kernel<true, false, false, false, false, true>
      <<<MROWS / 16, 256, 0, stream>>>(bank_h, Wq_t, bq, nullptr, nullptr, nullptr, q_h);
  // k = feat @ Wk + bk  (f16 out)
  gemm128_kernel<true, false, false, false, false, true>
      <<<NSROWS / 16, 256, 0, stream>>>(feat_h, Wk_t, bk, nullptr, nullptr, nullptr, k_h);
  // U = feat @ msg_W1[:128] + msg_b1
  gemm128_kernel<true, false, false, false, true, false>
      <<<NSROWS / 16, 256, 0, stream>>>(feat_h, W1a_t, msg_b1, nullptr, nullptr, U, nullptr);
  // V = bank @ msg_W1[128:256]
  gemm128_kernel<false, false, false, false, true, false>
      <<<MROWS / 16, 256, 0, stream>>>(bank_h, W1b_t, nullptr, nullptr, nullptr, V, nullptr);
  // P = feat @ upd_W1[:128] + upd_b1
  gemm128_kernel<true, false, false, false, true, false>
      <<<NSROWS / 16, 256, 0, stream>>>(feat_h, U1a_t, upd_b1, nullptr, nullptr, P, nullptr);

  // ---- stage 2: masked similarity + streaming top-20 + softmax ----
  sim_topk_kernel<<<NSROWS / 16, 256, 0, stream>>>(k_h, q_h, st, bst, topIdx, topScr);

  // ---- stage 3: message aggregation (linear-map factored through W2) ----
  msg_agg_kernel<<<NSROWS, CDIM, 0, stream>>>(U, V, bank_y, w1y, topIdx, topScr, G_h, ssum);

  // h_agg = G @ W2m + ssum*b2
  gemm128_kernel<true, true, false, false, false, true>
      <<<NSROWS / 16, 256, 0, stream>>>(G_h, W2m_t, msg_b2, ssum, nullptr, nullptr, H_h);
  // upd hidden = relu(P + h_agg @ U1b)      (upd_b1 already folded into P)
  gemm128_kernel<false, false, true, true, false, true>
      <<<NSROWS / 16, 256, 0, stream>>>(H_h, U1b_t, nullptr, nullptr, P, nullptr, R1_h);
  // out[:NS] = feat + hidden @ W2u + upd_b2
  gemm128_kernel<true, false, true, false, true, false>
      <<<NSROWS / 16, 256, 0, stream>>>(R1_h, W2u_t, upd_b2, nullptr, x, out, nullptr);

  // ---- stage 4: untouched tail rows of x ----
  {
    const int ntail = (NTROWS - NSROWS) * CDIM;
    copy_f32_kernel<<<(ntail + 255) / 256, 256, 0, stream>>>(
        x + (size_t)NSROWS * CDIM, out + (size_t)NSROWS * CDIM, ntail);
  }
}
